// VecKM_9620726743422
// MI455X (gfx1250) — matrix operand order repaired and checked by reading, not compiled, not run on hardware
//
#include <hip/hip_runtime.h>
#include <hip/hip_bf16.h>

typedef _Float16 v16h __attribute__((ext_vector_type(16)));
typedef _Float16 v8h  __attribute__((ext_vector_type(8)));
typedef _Float16 v4h  __attribute__((ext_vector_type(4)));
typedef float    v8f  __attribute__((ext_vector_type(8)));

#define NPTS   4096
#define DDIM   256
#define BETA2  40.5f
#define SQRT_D 16.0f

// ---------------------------------------------------------------------------
// Kernel 1: precompute transposed f16 tables cosT[b][d][m], sinT[b][d][m]
// (m contiguous so WMMA B-fragments are coalesced b128 loads).
// ---------------------------------------------------------------------------
__global__ __launch_bounds__(256) void veckm_build_tables(
    const float* __restrict__ pts, const float* __restrict__ A,
    _Float16* __restrict__ cosT, _Float16* __restrict__ sinT) {
  const int bd = blockIdx.x;          // 0..511
  const int b  = bd >> 8;
  const int d  = bd & 255;
  const float a0 = A[0 * DDIM + d];
  const float a1 = A[1 * DDIM + d];
  const float a2 = A[2 * DDIM + d];
  const float* pb = pts + (size_t)b * NPTS * 3;
  _Float16* cO = cosT + (size_t)bd * NPTS;
  _Float16* sO = sinT + (size_t)bd * NPTS;
  for (int m = threadIdx.x; m < NPTS; m += 256) {
    const float x = pb[m * 3 + 0], y = pb[m * 3 + 1], z = pb[m * 3 + 2];
    const float ph = x * a0 + y * a1 + z * a2;
    float s, c;
    __sincosf(ph, &s, &c);
    cO[m] = (_Float16)c;
    sO[m] = (_Float16)s;
  }
}

// ---------------------------------------------------------------------------
// Kernel 2: main WMMA kernel.
// Block = 256 threads = 8 waves; handles 16 rows (n-tile) x all 256 d.
// Wave w owns d columns 32w..32w+15 and 32w+16..32w+31 (Gr/Gi accumulators).
// m loop in chunks of 64: all threads build the shared 16x64 J tile in LDS
// (4 exp + one ds_store_b64 per thread), ONE barrier per chunk, then each
// wave reads 2 A-fragments (ds_load_b128 x4) and issues 8 WMMAs.
// ---------------------------------------------------------------------------
__global__ __launch_bounds__(256) void veckm_main(
    const float* __restrict__ pts, const float* __restrict__ A,
    const float* __restrict__ P,
    const _Float16* __restrict__ cosT, const _Float16* __restrict__ sinT,
    float* __restrict__ out) {
  __shared__ _Float16 Jbuf[2][16 * 64];   // [buf][row*64 + k], k = m - m0
  __shared__ float    pnBuf[16 * 3];      // the 16 n-rows of this block
  __shared__ float    rowNorm[16];

  const int tid  = threadIdx.x;
  const int lane = tid & 31;
  const int wave = tid >> 5;              // 0..7
  const int half = (lane >> 4) & 1;       // lane group
  const int lcol = lane & 15;

  const int bid = blockIdx.x;             // 0..511
  const int b   = bid >> 8;
  const int n0  = (bid & 255) << 4;

  const float* pb = pts + (size_t)b * NPTS * 3;

  // Stage the 16 row-points for epilogue use.
  if (tid < 48) pnBuf[tid] = pb[(size_t)n0 * 3 + tid];

  // J-builder assignment: thread (jr, jk4) computes J[jr][4*jk4 .. 4*jk4+3].
  const int jr  = tid & 15;
  const int jk4 = tid >> 4;               // 0..15 -> K base 4*jk4 in 0..63
  const float pnx = pb[(n0 + jr) * 3 + 0];
  const float pny = pb[(n0 + jr) * 3 + 1];
  const float pnz = pb[(n0 + jr) * 3 + 2];

  // Accumulators: 2 d-tiles x (real, imag).
  v8f accR0 = {}, accI0 = {}, accR1 = {}, accI1 = {};

  // Per-lane B-fragment base pointers (column = d, K runs along m).
  const int d0 = (wave << 5) + lcol;      // tile 0 column
  const int d1 = d0 + 16;                 // tile 1 column
  const int mlane = half << 4;            // lanes 0-15: K=0..15, lanes 16-31: K=16..31
  const _Float16* cos0 = cosT + (size_t)(b * DDIM + d0) * NPTS + mlane;
  const _Float16* sin0 = sinT + (size_t)(b * DDIM + d0) * NPTS + mlane;
  const _Float16* cos1 = cosT + (size_t)(b * DDIM + d1) * NPTS + mlane;
  const _Float16* sin1 = sinT + (size_t)(b * DDIM + d1) * NPTS + mlane;

  for (int m0 = 0; m0 < NPTS; m0 += 64) {
    const int buf = (m0 >> 6) & 1;

    // ---- build 16x64 J tile cooperatively (4 elements / thread) ----
    {
      const float* pm = pb + (size_t)(m0 + 4 * jk4) * 3;
      float e[4];
#pragma unroll
      for (int q = 0; q < 4; ++q) {
        const float dx = pnx - pm[3 * q + 0];
        const float dy = pny - pm[3 * q + 1];
        const float dz = pnz - pm[3 * q + 2];
        e[q] = __expf(-BETA2 * (dx * dx + dy * dy + dz * dz));
      }
      v4h jj;
      jj.x = (_Float16)e[0]; jj.y = (_Float16)e[1];
      jj.z = (_Float16)e[2]; jj.w = (_Float16)e[3];
      *(v4h*)&Jbuf[buf][jr * 64 + 4 * jk4] = jj;   // ds_store_b64
    }
    __syncthreads();                               // one barrier per 64-m chunk

    // ---- two K-subtiles of 32: A frags from LDS, B frags from tables ----
#pragma unroll
    for (int s = 0; s < 2; ++s) {
      const _Float16* jrow = &Jbuf[buf][(lane & 15) * 64 + 32 * s];
      const v8h alo = *(const v8h*)(jrow + 16 * half);       // K = 16h..16h+7
      const v8h ahi = *(const v8h*)(jrow + 16 * half + 8);   // K = 16h+8..16h+15
      const v16h aJ = __builtin_shufflevector(alo, ahi,
          0, 1, 2, 3, 4, 5, 6, 7, 8, 9, 10, 11, 12, 13, 14, 15);

      const int mo = m0 + 32 * s;
      const v16h bC0 = *(const v16h*)(cos0 + mo);
      const v16h bS0 = *(const v16h*)(sin0 + mo);
      const v16h bC1 = *(const v16h*)(cos1 + mo);
      const v16h bS1 = *(const v16h*)(sin1 + mo);

      accR0 = __builtin_amdgcn_wmma_f32_16x16x32_f16(false, aJ, false, bC0,
                                                     (short)0, accR0, false, false);
      accI0 = __builtin_amdgcn_wmma_f32_16x16x32_f16(false, aJ, false, bS0,
                                                     (short)0, accI0, false, false);
      accR1 = __builtin_amdgcn_wmma_f32_16x16x32_f16(false, aJ, false, bC1,
                                                     (short)0, accR1, false, false);
      accI1 = __builtin_amdgcn_wmma_f32_16x16x32_f16(false, aJ, false, bS1,
                                                     (short)0, accI1, false, false);
    }
  }

  // ------------------------- epilogue -------------------------
  __syncthreads();
  if (tid < 16) rowNorm[tid] = 0.0f;
  __syncthreads();

  // Per-lane A/P coefficients for the two owned d columns.
  float aC[2][3], pC[2][3];
#pragma unroll
  for (int t = 0; t < 2; ++t) {
    const int d = d0 + 16 * t;
    aC[t][0] = A[d]; aC[t][1] = A[DDIM + d]; aC[t][2] = A[2 * DDIM + d];
    pC[t][0] = P[d]; pC[t][1] = P[DDIM + d]; pC[t][2] = P[2 * DDIM + d];
  }

  // Rotate G by conj(e^{i phase}) and accumulate row norms.
  float GR[2][8], GI[2][8];
#pragma unroll
  for (int e = 0; e < 8; ++e) {
    const int r = e + 8 * half;           // C/D layout: VGPR e -> row e (+8 upper lanes)
    const float x = pnBuf[r * 3 + 0];
    const float y = pnBuf[r * 3 + 1];
    const float z = pnBuf[r * 3 + 2];
    float nrm = 0.0f;
#pragma unroll
    for (int t = 0; t < 2; ++t) {
      const float ph = x * aC[t][0] + y * aC[t][1] + z * aC[t][2];
      float s, c;
      __sincosf(ph, &s, &c);
      const float gr = t ? accR1[e] : accR0[e];
      const float gi = t ? accI1[e] : accI0[e];
      const float Gre = gr * c + gi * s;   // (Gr + iGi) * (c - i s)
      const float Gim = gi * c - gr * s;
      GR[t][e] = Gre; GI[t][e] = Gim;
      nrm += Gre * Gre + Gim * Gim;
    }
    atomicAdd(&rowNorm[r], nrm);          // ds_add_f32
  }
  __syncthreads();

  // Normalize, add e^{i pts.P}, store interleaved complex64.
  float2* outc = (float2*)out;
#pragma unroll
  for (int e = 0; e < 8; ++e) {
    const int r = e + 8 * half;
    const int n = n0 + r;
    const float x = pnBuf[r * 3 + 0];
    const float y = pnBuf[r * 3 + 1];
    const float z = pnBuf[r * 3 + 2];
    const float scale = SQRT_D * rsqrtf(rowNorm[r]);
#pragma unroll
    for (int t = 0; t < 2; ++t) {
      const int d = d0 + 16 * t;
      const float ph = x * pC[t][0] + y * pC[t][1] + z * pC[t][2];
      float s, c;
      __sincosf(ph, &s, &c);
      float2 o;
      o.x = GR[t][e] * scale + c;
      o.y = GI[t][e] * scale + s;
      outc[(size_t)(b * NPTS + n) * DDIM + d] = o;   // global_store_b64
    }
  }
}

// ---------------------------------------------------------------------------
extern "C" void kernel_launch(void* const* d_in, const int* in_sizes, int n_in,
                              void* d_out, int out_size, void* d_ws, size_t ws_size,
                              hipStream_t stream) {
  (void)in_sizes; (void)n_in; (void)out_size; (void)ws_size;
  const float* pts = (const float*)d_in[0];   // [2,4096,3]
  const float* A   = (const float*)d_in[1];   // [3,256]
  const float* P   = (const float*)d_in[2];   // [3,256]

  _Float16* cosT = (_Float16*)d_ws;                       // [2][256][4096] f16
  _Float16* sinT = cosT + (size_t)2 * DDIM * NPTS;        // +4 MB

  veckm_build_tables<<<dim3(2 * DDIM), dim3(256), 0, stream>>>(pts, A, cosT, sinT);
  veckm_main<<<dim3(2 * (NPTS / 16)), dim3(256), 0, stream>>>(
      pts, A, P, cosT, sinT, (float*)d_out);
}
